// CrossAttentionBlock_35098472743342
// MI455X (gfx1250) — compile-verified, matrix operand order checked by reading, not run on hardware
//
#include <hip/hip_runtime.h>
#include <hip/hip_bf16.h>

// ---------- types ----------
typedef __bf16 bf16;
typedef __attribute__((ext_vector_type(8)))  bf16  v8bf;
typedef __attribute__((ext_vector_type(16))) bf16  v16bf;
typedef __attribute__((ext_vector_type(8)))  float v8f;
typedef __attribute__((ext_vector_type(4)))  unsigned int v4u;
typedef __attribute__((ext_vector_type(8)))  int v8i;
typedef __attribute__((ext_vector_type(4)))  int v4i;

#define DIM   256
#define SEQ   4096
#define BATCH 8
#define NROW  (BATCH * SEQ)       // 32768
#define KTILES (SEQ / 32)         // 128

__device__ __forceinline__ v16bf load_frag(const bf16* p) {
  // 16-bit WMMA fragment: lo 8 elems at +0, hi 8 elems at +16 (ISA 7.12.2 layout)
  v8bf lo = *(const v8bf*)p;
  v8bf hi = *(const v8bf*)(p + 16);
  return __builtin_shufflevector(lo, hi, 0,1,2,3,4,5,6,7,8,9,10,11,12,13,14,15);
}

__device__ __forceinline__ v8f wmma_bf16(v16bf a, v16bf b, v8f c) {
  return __builtin_amdgcn_wmma_f32_16x16x32_bf16(false, a, false, b, (short)0, c,
                                                 false, false);
}

// ---------- TDM: 2D tile (32 rows x 256 cols, bf16) global -> LDS ----------
// D# per ISA 8.3/8.4: group0 = {flags, lds_addr, global_addr, type=2},
// group1 = {data_size=2B, tensor_dim0=256 (stride 256), tensor_dim1=4096,
//           tile_dim0=256, tile_dim1=32}. Groups 2/3 zero (2D tensor).
__device__ __forceinline__ void tdm_load_k_tile(const bf16* gsrc,
                                                unsigned lds_byte_off) {
  unsigned long long ga = (unsigned long long)(uintptr_t)gsrc;
  v4u g0;
  g0[0] = 1u;                                    // count=1, user descriptor
  g0[1] = lds_byte_off;                          // lds_addr
  g0[2] = (unsigned)(ga & 0xFFFFFFFFu);          // global_addr[31:0]
  g0[3] = (unsigned)((ga >> 32) & 0x01FFFFFFu)   // global_addr[56:32]
          | (2u << 30);                          // type = 2 ("image")
  v8i g1;
  g1[0] = (int)(1u << 16);                       // data_size = 1 -> 2 bytes
  g1[1] = (int)(256u << 16);                     // tensor_dim0 = 256
  g1[2] = (int)(4096u << 16);                    // tensor_dim1 = 4096
  g1[3] = (int)(256u << 16);                     // tile_dim0 = 256
  g1[4] = 32;                                    // tile_dim1 = 32
  g1[5] = 256;                                   // tensor_dim0_stride = 256
  g1[6] = 0;
  g1[7] = 0;
  v4i gz = {0, 0, 0, 0};
#if defined(__clang_major__) && (__clang_major__ >= 23)
  v8i gz8 = {0, 0, 0, 0, 0, 0, 0, 0};
  __builtin_amdgcn_tensor_load_to_lds(g0, g1, gz, gz, gz8, 0);
#else
  __builtin_amdgcn_tensor_load_to_lds(g0, g1, gz, gz, 0);
#endif
}

// ---------- kernel 0: fp32 -> bf16 conversion ----------
__global__ void f32_to_bf16_kernel(const float* __restrict__ src,
                                   bf16* __restrict__ dst, int n) {
  int i = (blockIdx.x * blockDim.x + threadIdx.x) * 8;
  if (i + 8 > n) return;
  float4 a = *(const float4*)(src + i);
  float4 b = *(const float4*)(src + i + 4);
  v8bf o;
  o[0] = (bf16)a.x; o[1] = (bf16)a.y; o[2] = (bf16)a.z; o[3] = (bf16)a.w;
  o[4] = (bf16)b.x; o[5] = (bf16)b.y; o[6] = (bf16)b.z; o[7] = (bf16)b.w;
  *(v8bf*)(dst + i) = o;
}

// ---------- kernel 1: QKV projection  y = x @ W^T + b  (q additionally * D^-0.5) ----------
// grid = (NROW/32, 3), block = 256 (8 waves). Each wave: 16 rows x 64 cols, K=256.
__global__ __launch_bounds__(256) void qkv_gemm_kernel(
    const bf16* __restrict__ xb, const bf16* __restrict__ wb,
    const float* __restrict__ bq, const float* __restrict__ bk,
    const float* __restrict__ bv,
    bf16* __restrict__ qo, bf16* __restrict__ ko, bf16* __restrict__ vo) {
  const int mat  = blockIdx.y;
  const bf16*  W    = wb + (size_t)mat * DIM * DIM;
  const float* bias = (mat == 0) ? bq : (mat == 1) ? bk : bv;
  bf16*        out  = (mat == 0) ? qo : (mat == 1) ? ko : vo;
  const float  scl  = (mat == 0) ? 0.0625f : 1.0f;  // 1/sqrt(256) folded into q

  const int w    = threadIdx.x >> 5;
  const int lane = threadIdx.x & 31;
  const int lr   = lane & 15, lh = lane >> 4;
  const int rowbase = blockIdx.x * 32 + (w >> 2) * 16;
  const int colbase = (w & 3) * 64;

  v16bf A[8];
  const bf16* xrow = xb + (size_t)(rowbase + lr) * DIM + lh * 8;
  #pragma unroll
  for (int kc = 0; kc < 8; ++kc) A[kc] = load_frag(xrow + kc * 32);

  v8f acc[4] = {{0,0,0,0,0,0,0,0},{0,0,0,0,0,0,0,0},
                {0,0,0,0,0,0,0,0},{0,0,0,0,0,0,0,0}};
  #pragma unroll
  for (int kc = 0; kc < 8; ++kc) {
    #pragma unroll
    for (int t = 0; t < 4; ++t) {
      const bf16* wrow = W + (size_t)(colbase + t * 16 + lr) * DIM + kc * 32 + lh * 8;
      v16bf Bf = load_frag(wrow);
      acc[t] = wmma_bf16(A[kc], Bf, acc[t]);
    }
  }
  #pragma unroll
  for (int t = 0; t < 4; ++t) {
    const int   col = colbase + t * 16 + lr;
    const float bb  = bias[col];
    #pragma unroll
    for (int i = 0; i < 8; ++i) {
      const int row = rowbase + i + 8 * lh;   // C layout: row = vgpr + 8*halflane
      out[(size_t)row * DIM + col] = (bf16)((acc[t][i] + bb) * scl);
    }
  }
}

// ---------- kernel 2: flash attention + residual (TDM + double buffer) ----------
// grid = (SEQ/128, BATCH), block = 256 (8 waves, 16 query rows each).
__global__ __launch_bounds__(256) void attn_kernel(
    const bf16* __restrict__ qb, const bf16* __restrict__ kb,
    const bf16* __restrict__ vb, const float* __restrict__ x,
    float* __restrict__ out) {
  __shared__ __align__(16) bf16 Kl[2][32 * DIM];   // K tiles, row-major [key][d]
  __shared__ __align__(16) bf16 Vt[2][DIM * 32];   // V tiles transposed [d][key]
  __shared__ __align__(16) bf16 Pl[8][16 * 32];    // per-wave P scratch

  const int batch = blockIdx.y;
  const int qbase = blockIdx.x * 128;
  const int tid   = threadIdx.x;
  const int w     = tid >> 5, lane = tid & 31;
  const int lr    = lane & 15, lh = lane >> 4;
  const size_t bo = (size_t)batch * SEQ * DIM;

  // Q fragments for this wave's 16 rows (scale already folded in)
  v16bf Qf[8];
  const bf16* qrow = qb + bo + (size_t)(qbase + w * 16 + lr) * DIM + lh * 8;
  #pragma unroll
  for (int kc = 0; kc < 8; ++kc) Qf[kc] = load_frag(qrow + kc * 32);

  v8f O[16];
  #pragma unroll
  for (int t = 0; t < 16; ++t)
    #pragma unroll
    for (int i = 0; i < 8; ++i) O[t][i] = 0.0f;
  float m[8], l[8];
  #pragma unroll
  for (int i = 0; i < 8; ++i) { m[i] = -3.0e38f; l[i] = 0.0f; }

  const int flat = tid * 32;
  const int krow = flat >> 8;      // 0..31   (key within tile)
  const int kcol = flat & 255;     // 0,32,.. (d offset)

  v8bf vr[4];   // in-flight V tile (register pipeline stage)
  auto load_v_regs = [&](int kt) {
    const bf16* vsrc = vb + bo + (size_t)(kt * 32 + krow) * DIM + kcol;
    #pragma unroll
    for (int j = 0; j < 4; ++j) vr[j] = *(const v8bf*)(vsrc + j * 8);
  };
  auto store_vt = [&](int buf) {
    #pragma unroll
    for (int j = 0; j < 4; ++j)
      #pragma unroll
      for (int e = 0; e < 8; ++e) Vt[buf][(kcol + j * 8 + e) * 32 + krow] = vr[j][e];
  };

  // ---- prologue: fill buffer 0 ----
  load_v_regs(0);
  if (tid < 32) tdm_load_k_tile(kb + bo, (unsigned)(uintptr_t)&Kl[0][0]);
  store_vt(0);
  if (tid < 32) __builtin_amdgcn_s_wait_tensorcnt(0);
  __syncthreads();

  for (int kt = 0; kt < KTILES; ++kt) {
    const int cur = kt & 1, nxt = cur ^ 1;
    const bool more = (kt + 1 < KTILES);

    // ---- kick off next tile: TDM for K, vector loads for V ----
    if (more) {
      load_v_regs(kt + 1);
      if (tid < 32)
        tdm_load_k_tile(kb + bo + (size_t)(kt + 1) * 32 * DIM,
                        (unsigned)(uintptr_t)&Kl[nxt][0]);
    }

    // ---- scores: S[16,32] = Q[16,256] @ K^T ----
    v8f s0 = {0,0,0,0,0,0,0,0}, s1 = {0,0,0,0,0,0,0,0};
    #pragma unroll
    for (int kc = 0; kc < 8; ++kc) {
      v16bf B0 = load_frag(&Kl[cur][(0  + lr) * DIM + kc * 32 + lh * 8]);
      v16bf B1 = load_frag(&Kl[cur][(16 + lr) * DIM + kc * 32 + lh * 8]);
      s0 = wmma_bf16(Qf[kc], B0, s0);
      s1 = wmma_bf16(Qf[kc], B1, s1);
    }

    // ---- online softmax (row = vgpr i + 8*lh; cols across 16-lane half) ----
    float corr[8];
    #pragma unroll
    for (int i = 0; i < 8; ++i) {
      float t = fmaxf(s0[i], s1[i]);
      t = fmaxf(t, __shfl_xor(t, 1, 32));
      t = fmaxf(t, __shfl_xor(t, 2, 32));
      t = fmaxf(t, __shfl_xor(t, 4, 32));
      t = fmaxf(t, __shfl_xor(t, 8, 32));
      const float mn = fmaxf(m[i], t);
      corr[i] = __expf(m[i] - mn);
      m[i] = mn;
      const float p0 = __expf(s0[i] - mn);
      const float p1 = __expf(s1[i] - mn);
      s0[i] = p0; s1[i] = p1;
      float r = p0 + p1;
      r += __shfl_xor(r, 1, 32);
      r += __shfl_xor(r, 2, 32);
      r += __shfl_xor(r, 4, 32);
      r += __shfl_xor(r, 8, 32);
      l[i] = l[i] * corr[i] + r;
    }
    #pragma unroll
    for (int t = 0; t < 16; ++t)
      #pragma unroll
      for (int i = 0; i < 8; ++i) O[t][i] *= corr[i];

    // ---- repack P: C layout -> A layout via per-wave LDS scratch ----
    #pragma unroll
    for (int i = 0; i < 8; ++i) {
      const int row = i + 8 * lh;
      Pl[w][row * 32 + lr]      = (bf16)s0[i];
      Pl[w][row * 32 + 16 + lr] = (bf16)s1[i];
    }
    v16bf Pf = load_frag(&Pl[w][lr * 32 + lh * 8]);  // same-wave DS: in order

    // ---- O[16,256] += P[16,32] @ V[32,256] ----
    #pragma unroll
    for (int t = 0; t < 16; ++t) {
      v16bf Bv = load_frag(&Vt[cur][(t * 16 + lr) * 32 + lh * 8]);
      O[t] = wmma_bf16(Pf, Bv, O[t]);
    }

    // ---- commit next tile into the other buffer ----
    if (more) {
      store_vt(nxt);
      if (tid < 32) __builtin_amdgcn_s_wait_tensorcnt(0);
    }
    __syncthreads();
  }

  // ---- epilogue: out = x + O / l ----
  float inv[8];
  #pragma unroll
  for (int i = 0; i < 8; ++i) inv[i] = 1.0f / l[i];
  #pragma unroll
  for (int t = 0; t < 16; ++t) {
    const int col = t * 16 + lr;
    #pragma unroll
    for (int i = 0; i < 8; ++i) {
      const int row = qbase + w * 16 + i + 8 * lh;
      const size_t idx = bo + (size_t)row * DIM + col;
      out[idx] = x[idx] + O[t][i] * inv[i];
    }
  }
}

// ---------- host ----------
extern "C" void kernel_launch(void* const* d_in, const int* in_sizes, int n_in,
                              void* d_out, int out_size, void* d_ws, size_t ws_size,
                              hipStream_t stream) {
  const float* x  = (const float*)d_in[0];
  const float* Wq = (const float*)d_in[1];
  const float* bq = (const float*)d_in[2];
  const float* Wk = (const float*)d_in[3];
  const float* bk = (const float*)d_in[4];
  const float* Wv = (const float*)d_in[5];
  const float* bv = (const float*)d_in[6];
  float* outp = (float*)d_out;

  const size_t NX = (size_t)NROW * DIM;   // 8388608
  const size_t NW = (size_t)DIM * DIM;    // 65536
  bf16* xb = (bf16*)d_ws;
  bf16* wb = xb + NX;
  bf16* qb = wb + 3 * NW;
  bf16* kb = qb + NX;
  bf16* vb = kb + NX;

  // fp32 -> bf16 staging
  f32_to_bf16_kernel<<<(int)(NX / 2048), 256, 0, stream>>>(x, xb, (int)NX);
  f32_to_bf16_kernel<<<(int)(NW / 2048), 256, 0, stream>>>(Wq, wb,          (int)NW);
  f32_to_bf16_kernel<<<(int)(NW / 2048), 256, 0, stream>>>(Wk, wb + NW,     (int)NW);
  f32_to_bf16_kernel<<<(int)(NW / 2048), 256, 0, stream>>>(Wv, wb + 2 * NW, (int)NW);

  // q/k/v projections (scale folded into q)
  qkv_gemm_kernel<<<dim3(NROW / 32, 3), 256, 0, stream>>>(xb, wb, bq, bk, bv,
                                                          qb, kb, vb);
  // flash attention + residual
  attn_kernel<<<dim3(SEQ / 128, BATCH), 256, 0, stream>>>(qb, kb, vb, x, outp);
}
